// MultiHeadDecoder_68118181314959
// MI455X (gfx1250) — matrix operand order repaired and checked by reading, not compiled, not run on hardware
//
#include <hip/hip_runtime.h>
#include <hip/hip_bf16.h>
#include <math.h>

typedef __attribute__((ext_vector_type(16))) _Float16 v16h;
typedef __attribute__((ext_vector_type(8)))  float    v8f;

#define B_  16
#define G_  384
#define D_  128
#define H_  4
#define KD_ 32
#define GT_ 24          // G_/16
#define VRANGE 6.0f

// K index mapping for the 16-bit A-matrix 16x32 VGPR layout:
// lanes 0-15 hold K in {0..7, 16..23}; lanes 16-31 hold K in {8..15, 24..31}
__device__ __forceinline__ int kmapA(int e, int hi) {
    return (e < 8) ? ((hi ? 8 : 0) + e) : ((hi ? 16 : 8) + e);
}

// ---------------------------------------------------------------------------
// Kernel 1: per-batch small projections of pickup/delivery rows + c_pd
// ---------------------------------------------------------------------------
__global__ void k1_batch_vecs(const float* __restrict__ h,
                              const float* __restrict__ Wq1,
                              const float* __restrict__ Wq2,
                              const float* __restrict__ Wk2,
                              const int* __restrict__ posp,
                              const int* __restrict__ posd,
                              float* __restrict__ Q1p, float* __restrict__ Q1d,
                              float* __restrict__ Q2p, float* __restrict__ Q2d,
                              float* __restrict__ K2p, float* __restrict__ K2d,
                              float* __restrict__ cpd)
{
    int b = blockIdx.x;
    int t = threadIdx.x;               // 0..127 => (head, k)
    __shared__ float hp[D_], hd[D_], s_q2p[H_ * KD_], s_k2d[H_ * KD_];
    int pp = posp[b], pd = posd[b];
    hp[t] = h[(size_t)(b * G_ + pp) * D_ + t];
    hd[t] = h[(size_t)(b * G_ + pd) * D_ + t];
    __syncthreads();
    int head = t / KD_, k = t % KD_;
    float aq1p = 0.f, aq1d = 0.f, aq2p = 0.f, aq2d = 0.f, ak2p = 0.f, ak2d = 0.f;
    for (int d = 0; d < D_; ++d) {
        float wq1 = Wq1[(head * D_ + d) * KD_ + k];
        float wq2 = Wq2[(head * D_ + d) * KD_ + k];
        float wk2 = Wk2[(head * D_ + d) * KD_ + k];
        aq1p += hp[d] * wq1; aq1d += hd[d] * wq1;
        aq2p += hp[d] * wq2; aq2d += hd[d] * wq2;
        ak2p += hp[d] * wk2; ak2d += hd[d] * wk2;
    }
    int o = b * H_ * KD_ + t;
    Q1p[o] = aq1p; Q1d[o] = aq1d; Q2p[o] = aq2p;
    Q2d[o] = aq2d; K2p[o] = ak2p; K2d[o] = ak2d;
    s_q2p[t] = aq2p; s_k2d[t] = ak2d;
    __syncthreads();
    if (t < H_) {
        float s = 0.f;
        for (int kk = 0; kk < KD_; ++kk) s += s_q2p[t * KD_ + kk] * s_k2d[t * KD_ + kk];
        cpd[b * H_ + t] = s;
    }
}

// ---------------------------------------------------------------------------
// Kernel 2: K1 = h @ Wk1, K2 = h @ Wk2 per head, via WMMA f16 (K=128 in 4 steps)
// One wave per (g-tile, batch). A = 16 rows of h (f16), B = Wk chunk 32x32.
// ---------------------------------------------------------------------------
__global__ void k2_proj_wmma(const float* __restrict__ h,
                             const float* __restrict__ Wk1,
                             const float* __restrict__ Wk2,
                             float* __restrict__ K1buf,
                             float* __restrict__ K2buf)
{
    int gt = blockIdx.x, b = blockIdx.y;
    int g0 = gt * 16;
    int lane = threadIdx.x;
    int hi = lane >> 4, r = lane & 15;

    // A tiles: 4 chunks of K=32 over the 128-dim embed; M row = g row = r
    v16h A[4];
    for (int kc = 0; kc < 4; ++kc)
        for (int e = 0; e < 16; ++e) {
            int K = kc * 32 + kmapA(e, hi);
            A[kc][e] = (_Float16)h[(size_t)(b * G_ + g0 + r) * D_ + K];
        }

    for (int w = 0; w < 2; ++w) {
        const float* W = w ? Wk2 : Wk1;
        float* dst = w ? K2buf : K1buf;
        for (int hh = 0; hh < H_; ++hh) {
            for (int nt = 0; nt < 2; ++nt) {
                v8f acc = {};
                for (int kc = 0; kc < 4; ++kc) {
                    // B 32x16: lanes = N, K in same order as A (kmapA)
                    v16h Bt;
                    for (int e = 0; e < 16; ++e) {
                        int K = kc * 32 + kmapA(e, hi);
                        Bt[e] = (_Float16)W[(hh * D_ + K) * KD_ + nt * 16 + r];
                    }
                    acc = __builtin_amdgcn_wmma_f32_16x16x32_f16(
                        false, A[kc], false, Bt, (short)0, acc, false, false);
                }
                // D: VGPR v -> M=v (lanes<16) / v+8 (lanes>=16), N=lane%16
                for (int vv = 0; vv < 8; ++vv) {
                    int g = g0 + vv + hi * 8;
                    dst[((size_t)(b * G_ + g) * H_ + hh) * KD_ + nt * 16 + r] = acc[vv];
                }
            }
        }
    }
}

// ---------------------------------------------------------------------------
// Kernel 3: per-(b,g) compat dots -> part_i, part_j, diagonal 'same' score
// ---------------------------------------------------------------------------
__global__ void k3_features(const float* __restrict__ K1buf,
                            const float* __restrict__ K2buf,
                            const float* __restrict__ Q1p, const float* __restrict__ Q1d,
                            const float* __restrict__ Q2p, const float* __restrict__ Q2d,
                            const float* __restrict__ K2p, const float* __restrict__ K2d,
                            const float* __restrict__ cpd,
                            const int* __restrict__ rec,
                            const float* __restrict__ W1, const float* __restrict__ b1,
                            const float* __restrict__ W2, const float* __restrict__ b2,
                            const float* __restrict__ w3, const float* __restrict__ b3,
                            float* __restrict__ parti, float* __restrict__ partj,
                            float* __restrict__ same_t)
{
    int tid = blockIdx.x * blockDim.x + threadIdx.x;
    if (tid >= B_ * G_) return;
    int b = tid / G_, g = tid % G_;
    float cpp[4], cdp[4], cppo[4], cdpo[4];
    const float* k1 = K1buf + (size_t)(b * G_ + g) * H_ * KD_;
    int rr = rec[b * G_ + g];
    const float* k2p_row = (rr == 0) ? (K2p + b * H_ * KD_)
                                     : (K2buf + (size_t)(b * G_ + rr) * H_ * KD_);
    const float* k2d_row = (rr == 0) ? (K2d + b * H_ * KD_)
                                     : (K2buf + (size_t)(b * G_ + rr) * H_ * KD_);
    for (int hh = 0; hh < H_; ++hh) {
        float s1 = 0.f, s2 = 0.f, s3 = 0.f, s4 = 0.f;
        for (int k = 0; k < KD_; ++k) {
            float kv = k1[hh * KD_ + k];
            s1 += Q1p[(b * H_ + hh) * KD_ + k] * kv;
            s2 += Q1d[(b * H_ + hh) * KD_ + k] * kv;
            s3 += Q2p[(b * H_ + hh) * KD_ + k] * k2p_row[hh * KD_ + k];
            s4 += Q2d[(b * H_ + hh) * KD_ + k] * k2d_row[hh * KD_ + k];
        }
        cpp[hh] = s1; cdp[hh] = s2; cppo[hh] = s3; cdpo[hh] = s4;
    }
    // fused fc1 halves: part_i = [cpp,cppo] @ W1[0:8], part_j = [cdp,cdpo] @ W1[8:16]
    for (int o = 0; o < 32; ++o) {
        float si = 0.f, sj = 0.f;
        for (int f = 0; f < 4; ++f) {
            si += cpp[f]  * W1[f * 32 + o]        + cppo[f] * W1[(4 + f) * 32 + o];
            sj += cdp[f]  * W1[(8 + f) * 32 + o]  + cdpo[f] * W1[(12 + f) * 32 + o];
        }
        parti[(size_t)(b * G_ + g) * 32 + o] = si;
        partj[(size_t)(b * G_ + g) * 32 + o] = sj;
    }
    // diagonal score: feat = (cpp, c_pd, 0, cdpo)
    float feat[16];
    for (int f = 0; f < 4; ++f) {
        feat[f] = cpp[f]; feat[4 + f] = cpd[b * H_ + f];
        feat[8 + f] = 0.f; feat[12 + f] = cdpo[f];
    }
    float hd1[32];
    for (int o = 0; o < 32; ++o) {
        float s = b1[o];
        for (int f = 0; f < 16; ++f) s += feat[f] * W1[f * 32 + o];
        hd1[o] = fmaxf(s, 0.f);
    }
    float acc = b3[0];
    for (int o = 0; o < 32; ++o) {
        float s = b2[o];
        for (int i2 = 0; i2 < 32; ++i2) s += hd1[i2] * W2[i2 * 32 + o];
        acc += fmaxf(s, 0.f) * w3[o];
    }
    same_t[b * G_ + g] = tanhf(acc) * VRANGE;
}

// ---------------------------------------------------------------------------
// Kernel 4: dominant pairwise MLP. One wave per 16x16 output tile.
// hid1 built in registers (never materialized); hid1 @ W2 via 2x WMMA per row;
// b2 injected via C operand; relu + w3 dot + 16-lane butterfly reduction.
// ---------------------------------------------------------------------------
__global__ void k4_pair_wmma(const float* __restrict__ parti,
                             const float* __restrict__ partj,
                             const float* __restrict__ same_t,
                             const float* __restrict__ W2,
                             const float* __restrict__ b1,
                             const float* __restrict__ b2,
                             const float* __restrict__ w3,
                             const float* __restrict__ b3,
                             float* __restrict__ out)
{
    int i0 = blockIdx.x * 16, j0 = blockIdx.y * 16, b = blockIdx.z;
    int lane = threadIdx.x;
    int hi = lane >> 4, r = lane & 15;
    __shared__ float lds_pi[16 * 32];
    __shared__ float lds_out[256];

    // cooperative load of the part_i tile (16 rows x 32 ch)
    for (int t = 0; t < 16; ++t) {
        int idx = lane * 16 + t;
        int row = idx >> 5, col = idx & 31;
        lds_pi[idx] = parti[(size_t)(b * G_ + i0 + row) * 32 + col];
    }
    // W2 as two f16 B-tiles (K=32, N=16 each)
    v16h W2b0, W2b1;
    for (int e = 0; e < 16; ++e) {
        int K = kmapA(e, hi);
        W2b0[e] = (_Float16)W2[K * 32 + r];
        W2b1[e] = (_Float16)W2[K * 32 + 16 + r];
    }
    float b2n0 = b2[r], b2n1 = b2[16 + r];
    float w3n0 = w3[r], w3n1 = w3[16 + r];
    float bias3 = b3[0];
    // part_j row for this lane's M-row (j = j0+r), pre-added with b1, in A K-order
    float pjb[16];
    for (int e = 0; e < 16; ++e) {
        int K = kmapA(e, hi);
        pjb[e] = partj[(size_t)(b * G_ + j0 + r) * 32 + K] + b1[K];
    }
    __syncthreads();

    v8f c0, c1;
    for (int vv = 0; vv < 8; ++vv) { c0[vv] = b2n0; c1[vv] = b2n1; }

    for (int mt = 0; mt < 16; ++mt) {
        // A row m (= j local) : hid1[i0+mt, j0+m, K] = relu(pi + pj + b1)
        v16h A;
        for (int e = 0; e < 16; ++e) {
            float v = lds_pi[mt * 32 + kmapA(e, hi)] + pjb[e];
            A[e] = (_Float16)fmaxf(v, 0.f);
        }
        v8f acc0 = __builtin_amdgcn_wmma_f32_16x16x32_f16(
            false, A, false, W2b0, (short)0, c0, false, false);
        v8f acc1 = __builtin_amdgcn_wmma_f32_16x16x32_f16(
            false, A, false, W2b1, (short)0, c1, false, false);
        // relu, * w3, reduce over the 32 output channels (16 lanes x 2 N-tiles)
        for (int vv = 0; vv < 8; ++vv) {
            float t = fmaxf(acc0[vv], 0.f) * w3n0 + fmaxf(acc1[vv], 0.f) * w3n1;
            t += __shfl_xor(t, 1); t += __shfl_xor(t, 2);
            t += __shfl_xor(t, 4); t += __shfl_xor(t, 8);
            if (r == 0) lds_out[mt * 16 + vv + hi * 8] = tanhf(t + bias3) * VRANGE;
        }
    }
    __syncthreads();
    // coalesced tile write with diagonal patch
    for (int t = 0; t < 8; ++t) {
        int idx = lane * 8 + t;
        int il = idx >> 4, jl = idx & 15;
        int gi = i0 + il, gj = j0 + jl;
        float v = lds_out[idx];
        if (gi == gj) v = same_t[b * G_ + gi];
        out[((size_t)(b * G_ + gi)) * G_ + gj] = v;
    }
}

// ---------------------------------------------------------------------------
extern "C" void kernel_launch(void* const* d_in, const int* in_sizes, int n_in,
                              void* d_out, int out_size, void* d_ws, size_t ws_size,
                              hipStream_t stream) {
    const float* h   = (const float*)d_in[0];
    const float* Wq1 = (const float*)d_in[1];
    const float* Wk1 = (const float*)d_in[2];
    const float* Wq2 = (const float*)d_in[3];
    const float* Wk2 = (const float*)d_in[4];
    const float* W1  = (const float*)d_in[5];
    const float* b1  = (const float*)d_in[6];
    const float* W2  = (const float*)d_in[7];
    const float* b2  = (const float*)d_in[8];
    const float* w3  = (const float*)d_in[9];
    const float* b3  = (const float*)d_in[10];
    const int* posp  = (const int*)d_in[11];
    const int* posd  = (const int*)d_in[12];
    const int* rec   = (const int*)d_in[13];
    float* out = (float*)d_out;

    float* ws = (float*)d_ws;
    size_t off = 0;
    float* K1buf = ws + off; off += (size_t)B_ * G_ * H_ * KD_;   // 786432
    float* K2buf = ws + off; off += (size_t)B_ * G_ * H_ * KD_;   // 786432
    float* Q1p   = ws + off; off += B_ * H_ * KD_;
    float* Q1d   = ws + off; off += B_ * H_ * KD_;
    float* Q2p   = ws + off; off += B_ * H_ * KD_;
    float* Q2d   = ws + off; off += B_ * H_ * KD_;
    float* K2p   = ws + off; off += B_ * H_ * KD_;
    float* K2d   = ws + off; off += B_ * H_ * KD_;
    float* cpd   = ws + off; off += B_ * H_;
    float* parti = ws + off; off += (size_t)B_ * G_ * 32;
    float* partj = ws + off; off += (size_t)B_ * G_ * 32;
    float* same_t= ws + off; off += B_ * G_;

    k1_batch_vecs<<<B_, 128, 0, stream>>>(h, Wq1, Wq2, Wk2, posp, posd,
                                          Q1p, Q1d, Q2p, Q2d, K2p, K2d, cpd);
    k2_proj_wmma<<<dim3(GT_, B_), 32, 0, stream>>>(h, Wk1, Wk2, K1buf, K2buf);
    k3_features<<<(B_ * G_ + 127) / 128, 128, 0, stream>>>(
        K1buf, K2buf, Q1p, Q1d, Q2p, Q2d, K2p, K2d, cpd, rec,
        W1, b1, W2, b2, w3, b3, parti, partj, same_t);
    k4_pair_wmma<<<dim3(GT_, GT_, B_), 32, 0, stream>>>(
        parti, partj, same_t, W2, b1, b2, w3, b3, out);
}
